// Head_65163243814913
// MI455X (gfx1250) — matrix operand order repaired and checked by reading, not compiled, not run on hardware
//
#include <hip/hip_runtime.h>

#define B_  8
#define S_  2048
#define E_  1024
#define HD_ 64

typedef unsigned short u16;
typedef __attribute__((ext_vector_type(16))) __bf16 v16bf;
typedef __attribute__((ext_vector_type(8)))  float  v8f;

union ABf {
    v16bf v;
    u16   u[16];
    uint4 q[2];
};

// single f32 -> bf16 by truncation (1 VALU op)
__device__ __forceinline__ u16 trunc_bf(float f) {
    return (u16)(__float_as_uint(f) >> 16);
}

// two f32 -> packed bf16x2 with one v_perm_b32
__device__ __forceinline__ unsigned pk2(float lo, float hi) {
    return __builtin_amdgcn_perm(__float_as_uint(hi), __float_as_uint(lo), 0x07060302u);
}

#define WMMA_BF16(A, Bm, C) \
    __builtin_amdgcn_wmma_f32_16x16x32_bf16(false, (A), false, (Bm), (short)0, (C), false, false)

// ---------------------------------------------------------------------------
// Kernel 0: one-time elementwise f32 -> bf16 staging.
// ---------------------------------------------------------------------------
__global__ __launch_bounds__(256) void cvt_bf16(const float* __restrict__ src,
                                                u16* __restrict__ dst, int n4) {
    int i = blockIdx.x * blockDim.x + threadIdx.x;
    if (i >= n4) return;
    float4 f = ((const float4*)src)[i];
    uint2 p;
    p.x = pk2(f.x, f.y);
    p.y = pk2(f.z, f.w);
    ((uint2*)dst)[i] = p;
}

// ---------------------------------------------------------------------------
// Kernel 1: Q/K/V projection, bf16 WMMA GEMM with software-pipelined
// (ping-pong) tile loads so WMMA overlaps VMEM latency.
// One wave -> one 16(seq) x 64(HD) tile of Q, K, or V.
// ---------------------------------------------------------------------------
__global__ __launch_bounds__(256) void qkv_proj(
    const u16* __restrict__ xbf, const u16* __restrict__ Wbf,  // Wq|Wk|Wv bf16
    u16* __restrict__ Qbf, u16* __restrict__ Kbf, u16* __restrict__ VT)
{
    const int lane = threadIdx.x & 31;
    const int wave = threadIdx.x >> 5;
    const int tile = blockIdx.x * 8 + wave;       // 0 .. 3*B*(S/16)-1
    const int mat  = tile % 3;                    // 0=Q, 1=K, 2=V
    const int rem  = tile / 3;
    const int b    = rem >> 7;                    // / (S/16)
    const int st   = rem & 127;
    const int half = lane >> 4;
    const int lm   = lane & 15;

    const int s = st * 16 + lm;                   // A-layout row M = lane%16
    const u16* xrow  = xbf + ((size_t)b * S_ + s) * E_;
    const u16* wbase = Wbf + (size_t)mat * HD_ * E_ + (size_t)lm * E_;

    auto loadA = [&](ABf& a, int ks) {            // A: K chunks 8h / 16+8h
        a.q[0] = *(const uint4*)(xrow + ks * 32 + half * 8);
        a.q[1] = *(const uint4*)(xrow + ks * 32 + 16 + half * 8);
    };
    auto loadB = [&](ABf* bw, int ks) {           // B: K chunks 8h / 16+8h
#pragma unroll
        for (int nt = 0; nt < 4; ++nt) {
            const u16* wp = wbase + (size_t)(nt * 16) * E_ + ks * 32 + half * 8;
            bw[nt].q[0] = *(const uint4*)(wp);
            bw[nt].q[1] = *(const uint4*)(wp + 16);
        }
    };

    ABf aA, aB, bwA[4], bwB[4];
    loadA(aA, 0);
    loadB(bwA, 0);

    v8f acc[4] = {};
    for (int ks = 0; ks < E_ / 32; ks += 2) {
        // prefetch odd step while computing even step
        loadA(aB, ks + 1);
        loadB(bwB, ks + 1);
#pragma unroll
        for (int nt = 0; nt < 4; ++nt) acc[nt] = WMMA_BF16(aA.v, bwA[nt].v, acc[nt]);
        // prefetch next even step while computing odd step
        if (ks + 2 < E_ / 32) {
            loadA(aA, ks + 2);
            loadB(bwA, ks + 2);
        }
#pragma unroll
        for (int nt = 0; nt < 4; ++nt) acc[nt] = WMMA_BF16(aB.v, bwB[nt].v, acc[nt]);
    }

    // C layout: M = r + 8*half, N = lane%16
    if (mat < 2) {
        u16* dst = (mat == 0) ? Qbf : Kbf;
#pragma unroll
        for (int nt = 0; nt < 4; ++nt)
#pragma unroll
            for (int r = 0; r < 8; ++r) {
                int M = r + 8 * half, N = nt * 16 + lm;
                dst[((size_t)b * S_ + st * 16 + M) * HD_ + N] = trunc_bf(acc[nt][r]);
            }
    } else {
#pragma unroll
        for (int nt = 0; nt < 4; ++nt)
#pragma unroll
            for (int r = 0; r < 8; ++r) {
                int M = r + 8 * half, N = nt * 16 + lm;
                VT[((size_t)b * HD_ + N) * S_ + st * 16 + M] = trunc_bf(acc[nt][r]);
            }
    }
}

// ---------------------------------------------------------------------------
// Kernel 2: causal flash attention, K tiles double-buffered across key-block
// iterations; V tiles issued before the score WMMAs so softmax hides their
// latency.  One wave -> 16 query rows.
// ---------------------------------------------------------------------------
__global__ __launch_bounds__(256) void attn(
    const u16* __restrict__ Qbf, const u16* __restrict__ Kbf,
    const u16* __restrict__ VT,  float* __restrict__ out)
{
    __shared__ u16 pbuf[8][16][32];               // per-wave P tile (bf16)

    const int lane = threadIdx.x & 31;
    const int wave = threadIdx.x >> 5;
    const int tile = blockIdx.x * 8 + wave;       // 0 .. B*(S/16)-1
    const int b    = tile >> 7;
    const int qt   = tile & 127;
    const int q0   = qt * 16;
    const int half = lane >> 4;
    const int lm   = lane & 15;

    const u16* kbase = Kbf + (size_t)b * S_ * HD_;
    const u16* vbase = VT  + (size_t)b * HD_ * S_;

    // Q as two A tiles (heads 0-31, 32-63), resident the whole kernel
    ABf aq0, aq1;
    {
        const u16* qrow = Qbf + ((size_t)b * S_ + q0 + lm) * HD_;
        aq0.q[0] = *(const uint4*)(qrow + half * 8);
        aq0.q[1] = *(const uint4*)(qrow + 16 + half * 8);
        aq1.q[0] = *(const uint4*)(qrow + 32 + half * 8);
        aq1.q[1] = *(const uint4*)(qrow + 48 + half * 8);
    }

    v8f   o[4] = {};
    float m_r[8], l_r[8];
#pragma unroll
    for (int r = 0; r < 8; ++r) { m_r[r] = -3.0e38f; l_r[r] = 0.f; }

    // kt[nt*2+ks]: B tiles for keys k0+16nt..k0+16nt+15, heads 32ks..32ks+31
    auto load_kt = [&](ABf* kt, int k0) {
#pragma unroll
        for (int nt = 0; nt < 2; ++nt)
#pragma unroll
            for (int ks = 0; ks < 2; ++ks) {
                const u16* kr = kbase + (size_t)(k0 + nt * 16 + lm) * HD_ + 32 * ks + half * 8;
                kt[nt * 2 + ks].q[0] = *(const uint4*)(kr);
                kt[nt * 2 + ks].q[1] = *(const uint4*)(kr + 16);
            }
    };

    auto block = [&](ABf* kt, int k0) {
        // V tiles early: latency hidden under score WMMAs + softmax
        ABf bv[4];
#pragma unroll
        for (int ht = 0; ht < 4; ++ht) {
            const u16* vrow = vbase + (size_t)(ht * 16 + lm) * S_ + k0 + half * 8;
            bv[ht].q[0] = *(const uint4*)(vrow);
            bv[ht].q[1] = *(const uint4*)(vrow + 16);
        }

        // ---- S = Q * K^T ----
        v8f s0 = {}, s1 = {};
        s0 = WMMA_BF16(aq0.v, kt[0].v, s0);
        s0 = WMMA_BF16(aq1.v, kt[1].v, s0);
        s1 = WMMA_BF16(aq0.v, kt[2].v, s1);
        s1 = WMMA_BF16(aq1.v, kt[3].v, s1);

        // ---- scale + causal mask + online softmax (row M = r+8*half) ----
        float alpha[8];
#pragma unroll
        for (int r = 0; r < 8; ++r) {
            const int M = r + 8 * half;
            const int q = q0 + M;
            float v0 = s0[r] * 0.125f;            // 1/sqrt(64)
            float v1 = s1[r] * 0.125f;
            if (k0 + lm > q)      v0 = -3.0e38f;  // key col N = lane%16
            if (k0 + 16 + lm > q) v1 = -3.0e38f;

            float mx = fmaxf(v0, v1);
#pragma unroll
            for (int msk = 1; msk < 16; msk <<= 1)
                mx = fmaxf(mx, __shfl_xor(mx, msk, 32));
            const float mnew = fmaxf(m_r[r], mx);

            const float p0 = __expf(v0 - mnew);
            const float p1 = __expf(v1 - mnew);
            float rs = p0 + p1;
#pragma unroll
            for (int msk = 1; msk < 16; msk <<= 1)
                rs += __shfl_xor(rs, msk, 32);

            alpha[r] = __expf(m_r[r] - mnew);
            l_r[r]   = l_r[r] * alpha[r] + rs;
            m_r[r]   = mnew;

            pbuf[wave][M][lm]      = trunc_bf(p0);  // C layout -> LDS
            pbuf[wave][M][16 + lm] = trunc_bf(p1);
        }

        asm volatile("s_wait_dscnt 0" ::: "memory");

        // reload P in A layout (row = lane%16, K chunks 8h / 16+8h)
        ABf pa;
        const u16* prow = &pbuf[wave][lm][0];
        pa.q[0] = *(const uint4*)(prow + half * 8);
        pa.q[1] = *(const uint4*)(prow + 16 + half * 8);

        // ---- O = alpha*O + P * V ----
#pragma unroll
        for (int ht = 0; ht < 4; ++ht)
#pragma unroll
            for (int r = 0; r < 8; ++r) o[ht][r] *= alpha[r];
#pragma unroll
        for (int ht = 0; ht < 4; ++ht)
            o[ht] = WMMA_BF16(pa.v, bv[ht].v, o[ht]);
    };

    const int nkb = (q0 + 16 + 31) >> 5;          // key blocks up to diagonal
    ABf ktA[4], ktB[4];
    load_kt(ktA, 0);

    int kb = 0;
    while (kb < nkb) {
        if (kb + 1 < nkb) load_kt(ktB, (kb + 1) * 32);   // prefetch next block
        block(ktA, kb * 32);
        ++kb;
        if (kb >= nkb) break;
        if (kb + 1 < nkb) load_kt(ktA, (kb + 1) * 32);   // prefetch next block
        block(ktB, kb * 32);
        ++kb;
    }

    // ---- normalize and store f32 [B,S,HD] ----
#pragma unroll
    for (int ht = 0; ht < 4; ++ht)
#pragma unroll
        for (int r = 0; r < 8; ++r) {
            const int M = r + 8 * half, N = ht * 16 + lm;
            out[((size_t)b * S_ + q0 + M) * HD_ + N] = o[ht][r] / l_r[r];
        }
}

extern "C" void kernel_launch(void* const* d_in, const int* in_sizes, int n_in,
                              void* d_out, int out_size, void* d_ws, size_t ws_size,
                              hipStream_t stream) {
    const float* x  = (const float*)d_in[0];
    const float* Wk = (const float*)d_in[1];
    const float* Wq = (const float*)d_in[2];
    const float* Wv = (const float*)d_in[3];
    float* out = (float*)d_out;

    const size_t QKV = (size_t)B_ * S_ * HD_;     // 1,048,576 elems each
    const size_t XN  = (size_t)B_ * S_ * E_;      // 16,777,216 elems
    const size_t WN  = (size_t)HD_ * E_;          // 65,536 elems each

    u16* ws  = (u16*)d_ws;
    u16* Qbf = ws;                                // 2 MB
    u16* Kbf = ws + QKV;                          // 2 MB
    u16* VT  = ws + 2 * QKV;                      // 2 MB
    u16* xbf = ws + 3 * QKV;                      // 32 MB
    u16* Wbf = ws + 3 * QKV + XN;                 // 384 KB (Wq | Wk | Wv)

    cvt_bf16<<<dim3((int)(XN / 4 + 255) / 256), dim3(256), 0, stream>>>(x, xbf, (int)(XN / 4));
    cvt_bf16<<<dim3((int)(WN / 4 + 255) / 256), dim3(256), 0, stream>>>(Wq, Wbf,          (int)(WN / 4));
    cvt_bf16<<<dim3((int)(WN / 4 + 255) / 256), dim3(256), 0, stream>>>(Wk, Wbf + WN,     (int)(WN / 4));
    cvt_bf16<<<dim3((int)(WN / 4 + 255) / 256), dim3(256), 0, stream>>>(Wv, Wbf + 2 * WN, (int)(WN / 4));

    qkv_proj<<<dim3(3 * B_ * (S_ / 16) / 8), dim3(256), 0, stream>>>(xbf, Wbf, Qbf, Kbf, VT);
    attn<<<dim3(B_ * (S_ / 16) / 8), dim3(256), 0, stream>>>(Qbf, Kbf, VT, out);
}
